// CausalAttention_66022237274192
// MI455X (gfx1250) — matrix operand order repaired and checked by reading, not compiled, not run on hardware
//
#include <hip/hip_runtime.h>
#include <math.h>

// ---------------------------------------------------------------------------
// Causal attention forward for MI455X (gfx1250, wave32, WMMA + TDM).
//   x:[4,2048,1024] f32, Wq/Wk/Wv:[1024,1024] f32 -> out:[4,2048,1024] f32
// All GEMMs: v_wmma_f32_16x16x32_f16, operands staged LDS via
// tensor_load_to_lds (TDM, double-buffered, TENSORcnt-gated).
// Workspace layout (MB):
//   [0,16)  x f16            [16,18) Wq^T f16   [18,20) Wk^T f16  [20,22) Wv^T f16
//   [22,38) Q f16            [38,54) K f16      [54,70) V^T f16
//   [70,134) scores f32
//   P f16 (32MB) aliases [22,54) -- Q/K dead when P is written.
// Total required ws: 134 MB.
// ---------------------------------------------------------------------------

typedef __attribute__((ext_vector_type(16))) _Float16 v16h;
typedef __attribute__((ext_vector_type(8)))  _Float16 v8h;
typedef __attribute__((ext_vector_type(8)))  float    v8f;
typedef __attribute__((ext_vector_type(4)))  unsigned int u32x4;
typedef __attribute__((ext_vector_type(8)))  int          i32x8;
typedef __attribute__((ext_vector_type(4)))  int          i32x4;

#define BATCH   4
#define SEQ     2048
#define DMODEL  1024

#define MT 128          // block tile M
#define NT 128          // block tile N
#define KT 32           // block tile K = one wmma k-step
#define LDSS 40         // LDS row stride in halves (32 + 8 pad): 80B rows, 16B aligned

__device__ __forceinline__ int imin(int a, int b) { return a < b ? a : b; }

// ---------------- TDM: DMA one [128 x 32] f16 tile into LDS -----------------
// K-major source rows of length kdim halves; LDS gets 4 DWORDs of pad after
// every 16 DWORDs (=32-half row), reproducing the LDSS=40 padded stride.
__device__ __forceinline__ void tdm_tile_load(const _Float16* gptr, unsigned lds_off,
                                              unsigned kdim) {
  unsigned long long ga = (unsigned long long)gptr;
  u32x4 g0;
  g0[0] = 1u;                                      // count=1, user descriptor
  g0[1] = lds_off;                                 // LDS byte address
  g0[2] = (unsigned)(ga & 0xffffffffu);            // global_addr[31:0]
  g0[3] = (unsigned)((ga >> 32) & 0x01ffffffu)     // global_addr[56:32]
        | (2u << 30);                              // type=2 (image)
  const unsigned TD1 = 1u << 20;                   // tensor_dim1 (OOB bound, unused)
  i32x8 g1;
  g1[0] = (int)((1u << 16)                         // data_size = 2 bytes
              | (1u << 20)                         // pad_enable
              | (3u << 22)                         // pad_interval: 16 DWORDs
              | (3u << 25));                       // pad_amount: 4 DWORDs
  g1[1] = (int)((kdim & 0xffffu) << 16);           // tensor_dim0[15:0]
  g1[2] = (int)((kdim >> 16) | ((TD1 & 0xffffu) << 16));
  g1[3] = (int)((TD1 >> 16) | ((unsigned)KT << 16)); // tile_dim0 = 32
  g1[4] = (int)MT;                                 // tile_dim1 = 128, tile_dim2 = 0
  g1[5] = (int)kdim;                               // tensor_dim0_stride[31:0]
  g1[6] = 0;                                       // stride[47:32] | dim1_stride lo
  g1[7] = 0;
  const i32x4 z4 = {0, 0, 0, 0};
  const i32x8 z8 = {0, 0, 0, 0, 0, 0, 0, 0};
  __builtin_amdgcn_tensor_load_to_lds(g0, g1, z4, z4, z8, 0);  // 6-arg form
}

__device__ __forceinline__ unsigned lds_addr_of(const void* p) {
  return (unsigned)(unsigned long long)p;          // low 32 bits = LDS offset
}

// A fragment (16x32 f16, M x K). ISA layout:
//  lanes 0-15 : row M=lane,  halves 0..7 = K 0..7,  halves 8..15 = K 16..23
//  lanes 16-31: row M=lane-16, halves 0..7 = K 8..15, halves 8..15 = K 24..31
__device__ __forceinline__ v16h load_a_frag(const _Float16* tile, int row, int lane) {
  const int up  = (lane & 16) ? 1 : 0;
  const _Float16* p = tile + row * LDSS;
  v8h lo = *(const v8h*)(p + (up ? 8 : 0));
  v8h hi = *(const v8h*)(p + (up ? 24 : 16));
  return __builtin_shufflevector(lo, hi, 0,1,2,3,4,5,6,7,8,9,10,11,12,13,14,15);
}

// B fragment (32x16 f16, K x N), LDS is K-major per column:
//  lanes 0-15: col N=lane, halves K=0..7,16..23 ; lanes 16-31: col N=lane-16, K=8..15,24..31
__device__ __forceinline__ v16h load_b_frag(const _Float16* tile, int col, int lane) {
  const _Float16* p = tile + col * LDSS + ((lane & 16) ? 8 : 0);
  v8h lo = *(const v8h*)(p);
  v8h hi = *(const v8h*)(p + 16);
  return __builtin_shufflevector(lo, hi, 0,1,2,3,4,5,6,7,8,9,10,11,12,13,14,15);
}

// ------------------------- fp32 -> f16 convert ------------------------------
__global__ __launch_bounds__(256) void cvt_f32_f16(const float* __restrict__ in,
                                                   _Float16* __restrict__ out, int n) {
  int i = blockIdx.x * 256 + threadIdx.x;
  if (i < n) out[i] = (_Float16)in[i];
}

// ------------- fp32 -> f16 transposing convert (1024x1024) ------------------
__global__ __launch_bounds__(256) void cvt_transpose_f32_f16(const float* __restrict__ in,
                                                             _Float16* __restrict__ out) {
  __shared__ float tile[32][33];
  const int bx = blockIdx.x * 32, by = blockIdx.y * 32;
  const int tx = threadIdx.x & 31, ty = threadIdx.x >> 5;   // 32 x 8
  for (int r = ty; r < 32; r += 8)
    tile[r][tx] = in[(long)(by + r) * DMODEL + bx + tx];
  __syncthreads();
  for (int r = ty; r < 32; r += 8)
    out[(long)(bx + r) * DMODEL + by + tx] = (_Float16)tile[tx][r];
}

// --------------- unified tiled WMMA GEMM: C = A @ Bt^T ----------------------
// A : [M, Kdim]   f16 (row stride Kdim)      -- K-major
// Bt: [Ncols, Kdim] f16 (row stride Kdim)    -- K-major ("B transposed")
// C : [M, Ncols] (CT); grid.z = batch via given strides.
// CAUSAL_K   : clamp k loop at the causal diagonal   (O = P @ V)
// CAUSAL_SKIP: drop tiles above the diagonal         (scores)
// TRANS_C    : write C transposed per batch          (V projection)
template <typename CT, bool CAUSAL_K, bool CAUSAL_SKIP, bool TRANS_C>
__global__ __launch_bounds__(256)
void gemm_wmma_tdm(const _Float16* __restrict__ Ag, const _Float16* __restrict__ Bg,
                   CT* __restrict__ Cg, int Kdim, int Ncols,
                   long aBatch, long bBatch, long cBatch) {
  __shared__ __align__(16) _Float16 As[2][MT * LDSS];
  __shared__ __align__(16) _Float16 Bs[2][NT * LDSS];

  const int i0 = blockIdx.x * MT;
  const int n0 = blockIdx.y * NT;
  if (CAUSAL_SKIP && n0 > i0 + MT) return;     // tile fully above j<=i+1 mask

  const int tid  = threadIdx.x;
  const int lane = tid & 31;
  const int wid  = tid >> 5;
  const int wm   = wid >> 1;                   // 0..3 -> 32-row slice
  const int wn   = wid & 1;                    // 0..1 -> 64-col slice
  const int b    = blockIdx.z;
  const _Float16* A  = Ag + (long)b * aBatch;
  const _Float16* Bt = Bg + (long)b * bBatch;
  CT*             C  = Cg + (long)b * cBatch;

  int kmax = Kdim;
  if (CAUSAL_K) {                              // rows i0..i0+127 need k <= i0+MT
    int lim = ((i0 + MT + 1 + KT - 1) / KT) * KT;
    kmax = imin(Kdim, lim);
  }
  const int ntiles = kmax / KT;

  const v8f vzero = {0.f, 0.f, 0.f, 0.f, 0.f, 0.f, 0.f, 0.f};
  v8f acc[2][4];
#pragma unroll
  for (int sm = 0; sm < 2; ++sm)
#pragma unroll
    for (int sn = 0; sn < 4; ++sn) acc[sm][sn] = vzero;

  auto issue = [&](int t) {
    const int bufi = t & 1;
    const int k0 = t * KT;
    tdm_tile_load(A  + (long)i0 * Kdim + k0, lds_addr_of(&As[bufi][0]), (unsigned)Kdim);
    tdm_tile_load(Bt + (long)n0 * Kdim + k0, lds_addr_of(&Bs[bufi][0]), (unsigned)Kdim);
  };

  if (wid == 0) issue(0);                      // prime the pipeline (wave 0 only)

  for (int t = 0; t < ntiles; ++t) {
    if (wid == 0) {
      if (t + 1 < ntiles) {
        issue(t + 1);                          // prefetch next tile into other buffer
        __builtin_amdgcn_s_wait_tensorcnt(2);  // oldest 2 ops (tile t) complete
      } else {
        __builtin_amdgcn_s_wait_tensorcnt(0);
      }
    }
    __syncthreads();                           // tile t visible to all waves

    const _Float16* Ac = As[t & 1];
    const _Float16* Bc = Bs[t & 1];
    v16h af[2], bf[4];
#pragma unroll
    for (int sm = 0; sm < 2; ++sm)
      af[sm] = load_a_frag(Ac, wm * 32 + sm * 16 + (lane & 15), lane);
#pragma unroll
    for (int sn = 0; sn < 4; ++sn)
      bf[sn] = load_b_frag(Bc, wn * 64 + sn * 16 + (lane & 15), lane);
#pragma unroll
    for (int sm = 0; sm < 2; ++sm)
#pragma unroll
      for (int sn = 0; sn < 4; ++sn)
        acc[sm][sn] = __builtin_amdgcn_wmma_f32_16x16x32_f16(
            false, af[sm], false, bf[sn], (short)0, acc[sm][sn], false, false);

    __syncthreads();                           // done reading buf before reuse
  }

  // store per C/D layout: VGPR e -> row base+e (lanes<16) / base+e+8 (lanes>=16)
  const int rsub = (lane & 16) ? 8 : 0;
  const int csub = lane & 15;
#pragma unroll
  for (int sm = 0; sm < 2; ++sm) {
    const int rbase = i0 + wm * 32 + sm * 16 + rsub;
#pragma unroll
    for (int sn = 0; sn < 4; ++sn) {
      const int col = n0 + wn * 64 + sn * 16 + csub;
#pragma unroll
      for (int e = 0; e < 8; ++e) {
        if (TRANS_C) {                         // V^T: [batch][Ncols][SEQ]
          const long r = rbase + e;            // global row over all batches
          const long bb = r >> 11, ii = r & (SEQ - 1);
          Cg[(bb * Ncols + col) * SEQ + ii] = (CT)acc[sm][sn][e];
        } else {
          C[(long)(rbase + e) * Ncols + col] = (CT)acc[sm][sn][e];
        }
      }
    }
  }
}

// -------- row softmax of scores/32 with causal mask; P written in f16 -------
__global__ __launch_bounds__(256)
void softmax_kernel(const float* __restrict__ Sc, _Float16* __restrict__ P) {
  __shared__ float red[256];
  const int row = blockIdx.x;                  // 0 .. BATCH*SEQ-1
  const int b   = row >> 11;
  const int i   = row & (SEQ - 1);
  const float* s = Sc + ((long)b * SEQ + i) * SEQ;
  _Float16*    p = P  + ((long)b * SEQ + i) * SEQ;
  const int nvalid = imin(i + 2, SEQ);         // mask: j <= i+1
  const float NEG = -1.0e30f;
  const float scale = 0.03125f;                // 1/sqrt(1024)

  float vals[SEQ / 256];
  float m = NEG;
#pragma unroll
  for (int t = 0; t < SEQ / 256; ++t) {
    int j = threadIdx.x + t * 256;
    float v = (j < nvalid) ? s[j] * scale : NEG;
    vals[t] = v;
    m = fmaxf(m, v);
  }
  red[threadIdx.x] = m;
  __syncthreads();
  for (int off = 128; off > 0; off >>= 1) {
    if (threadIdx.x < off) red[threadIdx.x] = fmaxf(red[threadIdx.x], red[threadIdx.x + off]);
    __syncthreads();
  }
  m = red[0];
  __syncthreads();

  float sum = 0.f;
#pragma unroll
  for (int t = 0; t < SEQ / 256; ++t) {
    float e = __expf(vals[t] - m);             // masked: exp(-huge) -> 0
    vals[t] = e;
    sum += e;
  }
  red[threadIdx.x] = sum;
  __syncthreads();
  for (int off = 128; off > 0; off >>= 1) {
    if (threadIdx.x < off) red[threadIdx.x] += red[threadIdx.x + off];
    __syncthreads();
  }
  const float inv = 1.0f / red[0];
#pragma unroll
  for (int t = 0; t < SEQ / 256; ++t) {
    int j = threadIdx.x + t * 256;
    p[j] = (_Float16)(vals[t] * inv);          // zeros above mask, full row written
  }
}

// ---------------------------------------------------------------------------
extern "C" void kernel_launch(void* const* d_in, const int* in_sizes, int n_in,
                              void* d_out, int out_size, void* d_ws, size_t ws_size,
                              hipStream_t stream) {
  (void)in_sizes; (void)n_in; (void)out_size; (void)ws_size;
  const float* x  = (const float*)d_in[0];
  const float* Wq = (const float*)d_in[1];
  const float* Wk = (const float*)d_in[2];
  const float* Wv = (const float*)d_in[3];
  float* out = (float*)d_out;

  char* ws = (char*)d_ws;
  const size_t MB = (size_t)1 << 20;
  _Float16* xh  = (_Float16*)(ws + 0 * MB);    // 16 MB
  _Float16* Wqt = (_Float16*)(ws + 16 * MB);
  _Float16* Wkt = (_Float16*)(ws + 18 * MB);
  _Float16* Wvt = (_Float16*)(ws + 20 * MB);
  _Float16* Qh  = (_Float16*)(ws + 22 * MB);   // 16 MB
  _Float16* Kh  = (_Float16*)(ws + 38 * MB);   // 16 MB
  _Float16* Vt  = (_Float16*)(ws + 54 * MB);   // 16 MB, [B][1024][2048]
  float*    Sc  = (float*)   (ws + 70 * MB);   // 64 MB
  _Float16* P   = (_Float16*)(ws + 22 * MB);   // aliases dead Q/K region

  // f32 -> f16 conversions
  const int NX = BATCH * SEQ * DMODEL;         // 8M
  cvt_f32_f16<<<dim3(NX / 256), dim3(256), 0, stream>>>(x, xh, NX);
  dim3 gt(DMODEL / 32, DMODEL / 32);
  cvt_transpose_f32_f16<<<gt, dim3(256), 0, stream>>>(Wq, Wqt);
  cvt_transpose_f32_f16<<<gt, dim3(256), 0, stream>>>(Wk, Wkt);
  cvt_transpose_f32_f16<<<gt, dim3(256), 0, stream>>>(Wv, Wvt);

  // Projections: Q/K normal, V transposed store
  dim3 gp(BATCH * SEQ / MT, DMODEL / NT, 1);
  gemm_wmma_tdm<_Float16, false, false, false><<<gp, dim3(256), 0, stream>>>(
      xh, Wqt, Qh, DMODEL, DMODEL, 0, 0, 0);
  gemm_wmma_tdm<_Float16, false, false, false><<<gp, dim3(256), 0, stream>>>(
      xh, Wkt, Kh, DMODEL, DMODEL, 0, 0, 0);
  gemm_wmma_tdm<_Float16, false, false, true><<<gp, dim3(256), 0, stream>>>(
      xh, Wvt, Vt, DMODEL, DMODEL, 0, 0, 0);

  // Causal scores: S = Q @ K^T (upper tiles skipped in-kernel)
  dim3 gs(SEQ / MT, SEQ / NT, BATCH);
  gemm_wmma_tdm<float, false, true, false><<<gs, dim3(256), 0, stream>>>(
      Qh, Kh, Sc, DMODEL, SEQ,
      (long)SEQ * DMODEL, (long)SEQ * DMODEL, (long)SEQ * SEQ);

  // Row softmax -> P (f16, zeros above mask)
  softmax_kernel<<<dim3(BATCH * SEQ), dim3(256), 0, stream>>>(Sc, P);

  // O = P @ V with causal k-limit
  dim3 go(SEQ / MT, DMODEL / NT, BATCH);
  gemm_wmma_tdm<float, true, false, false><<<go, dim3(256), 0, stream>>>(
      P, Vt, out, SEQ, DMODEL,
      (long)SEQ * SEQ, (long)DMODEL * SEQ, (long)SEQ * DMODEL);
}
